// DIEN_34935263986148
// MI455X (gfx1250) — compile-verified, matrix operand order checked by reading, not run on hardware
//
#include <hip/hip_runtime.h>
#include <hip/hip_bf16.h>

typedef __attribute__((ext_vector_type(16))) _Float16 v16h;
typedef __attribute__((ext_vector_type(8)))  float    v8f;
typedef __attribute__((ext_vector_type(4)))  unsigned v4u;

#define Bc 1024
#define Tc 200
#define Dc 128
#define Hc 128
#define KD 256          /* D+H */
#define NG 256          /* 2H  */
#define W_PITCH 264     /* f16 pitch, 16B-aligned rows, 132-dword stride (bank-conflict-free) */
#define F_PITCH 132     /* f32 pitch */
#define NTHREADS 256

/* LDS layout (bytes) — dynamic LDS starts at offset 0 (no static __shared__ used) */
#define OFF_WGT   0
#define SZ_WGT    (NG * W_PITCH * 2)            /* 135168: W_gate^T f16 */
#define OFF_WCT   (OFF_WGT + SZ_WGT)
#define SZ_WCT    (Hc * W_PITCH * 2)            /*  67584: W_cand^T f16 */
#define OFF_A     (OFF_WCT + SZ_WCT)
#define SZ_A      (16 * W_PITCH * 2)            /*   8448: A = [x|h] f16 */
#define OFF_H     (OFF_A + SZ_A)
#define SZ_F      (16 * F_PITCH * 4)            /*   8448 each */
#define OFF_R     (OFF_H + SZ_F)
#define OFF_U     (OFF_R + SZ_F)
#define OFF_SLEN  (OFF_U + SZ_F)
#define OFF_XS    (OFF_SLEN + 64)               /* double-buffered x stage, f32 [2][16][128] */
#define SZ_XS     (2 * 16 * Dc * 4)             /*  16384 */
#define SMEM_BYTES (OFF_XS + SZ_XS)             /* 252992 < 320KB */

union FragU { v16h h; v4u q[2]; };

/* 16-bit A/B fragment for 16x16x32 WMMA from row-major f16 LDS (pitch W_PITCH).
   VGPR v0..3 = K {0,2,4,6}+khalf*8 pairs, v4..7 = +16. Two b128 loads, 16B aligned. */
__device__ __forceinline__ v16h load_frag(const _Float16* base, int row, int kc, int khalf) {
    const int off = row * W_PITCH + kc * 32 + khalf * 8;
    FragU f;
    f.q[0] = *(const v4u*)(base + off);
    f.q[1] = *(const v4u*)(base + off + 16);
    return f.h;
}

__device__ __forceinline__ float fast_sigmoid(float x) {
    return 1.0f / (1.0f + __expf(-x));
}
__device__ __forceinline__ float fast_tanh(float x) {
    float e = __expf(2.0f * x);
    return (e - 1.0f) / (e + 1.0f);
}

/* Async DMA of one x timestep tile (16 rows x 128 f32 = 8KB) into LDS stage.
   Each of the 256 threads issues two GLOBAL_LOAD_ASYNC_TO_LDS_B128 (GVS mode:
   SGPR 64-bit base + 32-bit VGPR byte offset). Tracked by ASYNCcnt. */
__device__ __forceinline__ void prefetch_x(const float* x, int rowBase, int gt,
                                           int stage, int tid) {
#pragma unroll
    for (int j = 0; j < 2; ++j) {
        int c = tid + j * NTHREADS;                 /* 16B chunk id, 0..511 */
        int m = c >> 5;                             /* batch row in tile    */
        int d = (c & 31) * 4;                       /* feature base         */
        unsigned goff = (unsigned)((((unsigned)(rowBase + m) * Tc + (unsigned)gt) * Dc + d) * 4u);
        unsigned loff = (unsigned)(OFF_XS + stage * (16 * Dc * 4) + c * 16);
        asm volatile("global_load_async_to_lds_b128 %0, %1, %2"
                     :: "v"(loff), "v"(goff),
                        "s"((unsigned long long)(uintptr_t)x)
                     : "memory");
    }
}

__global__ void __launch_bounds__(NTHREADS, 1)
dien_gru_kernel(const float* __restrict__ x,       /* [B,T,D] */
                const int*   __restrict__ seq_len, /* [B]     */
                const float* __restrict__ Wg,      /* [256,256] */
                const float* __restrict__ bg,      /* [256]   */
                const float* __restrict__ Wc,      /* [256,128] */
                const float* __restrict__ bc,      /* [128]   */
                float*       __restrict__ out)     /* [B,T,H] */
{
    extern __shared__ char smem[];
    _Float16* WgT   = (_Float16*)(smem + OFF_WGT);
    _Float16* WcT   = (_Float16*)(smem + OFF_WCT);
    _Float16* A     = (_Float16*)(smem + OFF_A);
    float*    h_lds = (float*)(smem + OFF_H);
    float*    r_lds = (float*)(smem + OFF_R);
    float*    u_lds = (float*)(smem + OFF_U);
    int*      slen  = (int*)(smem + OFF_SLEN);
    const float* xs = (const float*)(smem + OFF_XS);

    const int tid     = threadIdx.x;
    const int lane    = tid & 31;
    const int wave    = tid >> 5;
    const int rowBase = blockIdx.x * 16;

    /* kick off DMA for t=0 while we fill weights */
    prefetch_x(x, rowBase, 0, 0, tid);

    /* ---- one-time init: weights -> LDS (transposed, f16), h=0, seq_len ---- */
    for (int i = tid; i < KD * NG; i += NTHREADS) {
        int k = i >> 8, n = i & 255;
        WgT[n * W_PITCH + k] = (_Float16)Wg[i];
    }
    for (int i = tid; i < KD * Hc; i += NTHREADS) {
        int k = i >> 7, n = i & 127;
        WcT[n * W_PITCH + k] = (_Float16)Wc[i];
    }
    for (int i = tid; i < 16 * F_PITCH; i += NTHREADS) h_lds[i] = 0.0f;
    if (tid < 16) slen[tid] = seq_len[rowBase + tid];
    __syncthreads();

    /* per-wave tile assignment */
    const int nb0 = wave * 32;        /* gate tile 0 */
    const int nb1 = wave * 32 + 16;   /* gate tile 1 */
    const int ncd = wave * 16;        /* cand tile   */
    const int nl  = lane & 15;
    const int mb  = (lane >> 4) << 3; /* 0 or 8 */
    const int am  = lane & 15;        /* A row for this lane */
    const int khalf = lane >> 4;

    const float bg0 = bg[nb0 + nl];
    const float bg1 = bg[nb1 + nl];
    const float bcv = bc[ncd + nl];

    int slv[8];
#pragma unroll
    for (int v = 0; v < 8; ++v) slv[v] = slen[mb + v];

    /* W_cand is loop-invariant: keep all 8 B fragments in registers (64 VGPRs) */
    v16h bcf[8];
#pragma unroll
    for (int kc = 0; kc < 8; ++kc) bcf[kc] = load_frag(WcT, ncd + nl, kc, khalf);

    for (int t = 0; t < Tc; ++t) {
        /* DMA next step's x tile into the other stage (clamp addr on last step) */
        const int gt = (t + 1 < Tc) ? t + 1 : Tc - 1;
        prefetch_x(x, rowBase, gt, (t + 1) & 1, tid);
        /* loads complete in order: <=2 outstanding guarantees stage[t&1] resident */
        asm volatile("s_wait_asynccnt 0x2" ::: "memory");
        __syncthreads();   /* stage visible to all waves; h_lds from prev combine */

        /* ---- build A = [x_t | h] in f16 (x from LDS stage) ---- */
        for (int i = tid; i < 16 * Dc; i += NTHREADS) {
            int m = i >> 7, d = i & 127;
            A[m * W_PITCH + d]       = (_Float16)xs[(t & 1) * (16 * Dc) + i];
            A[m * W_PITCH + 128 + d] = (_Float16)h_lds[m * F_PITCH + d];
        }
        __syncthreads();

        /* ---- gates: [16x256] @ [256x256], two N-tiles per wave ---- */
        v8f g0 = { bg0, bg0, bg0, bg0, bg0, bg0, bg0, bg0 };
        v8f g1 = { bg1, bg1, bg1, bg1, bg1, bg1, bg1, bg1 };
        v16h ax[4];   /* cache x-half A fragments (kc 0..3) for the cand matmul */
#pragma unroll
        for (int kc = 0; kc < 8; ++kc) {
            v16h a  = load_frag(A,   am,       kc, khalf);
            if (kc < 4) ax[kc] = a;
            v16h b0 = load_frag(WgT, nb0 + nl, kc, khalf);
            v16h b1 = load_frag(WgT, nb1 + nl, kc, khalf);
            g0 = __builtin_amdgcn_wmma_f32_16x16x32_f16(false, a, false, b0, (short)0, g0, false, false);
            g1 = __builtin_amdgcn_wmma_f32_16x16x32_f16(false, a, false, b1, (short)0, g1, false, false);
        }

        /* sigmoid -> r (cols 0..127) / u (cols 128..255); nb uniform per wave */
        {
            float* dst0 = (nb0 < 128) ? (r_lds + nb0 + nl) : (u_lds + (nb0 - 128) + nl);
#pragma unroll
            for (int v = 0; v < 8; ++v) dst0[(mb + v) * F_PITCH] = fast_sigmoid(g0[v]);
            float* dst1 = (nb1 < 128) ? (r_lds + nb1 + nl) : (u_lds + (nb1 - 128) + nl);
#pragma unroll
            for (int v = 0; v < 8; ++v) dst1[(mb + v) * F_PITCH] = fast_sigmoid(g1[v]);
        }
        __syncthreads();

        /* ---- rebuild A tail with r*h (cols 128..255 only; x half untouched) ---- */
        for (int i = tid; i < 16 * Hc; i += NTHREADS) {
            int m = i >> 7, d = i & 127;
            A[m * W_PITCH + 128 + d] =
                (_Float16)(r_lds[m * F_PITCH + d] * h_lds[m * F_PITCH + d]);
        }
        __syncthreads();

        /* ---- candidate: [16x256] @ [256x128]; B from regs, A x-half from regs ---- */
        v8f cA = { bcv, bcv, bcv, bcv, bcv, bcv, bcv, bcv };
#pragma unroll
        for (int kc = 0; kc < 8; ++kc) {
            v16h a = (kc < 4) ? ax[kc] : load_frag(A, am, kc, khalf);
            cA = __builtin_amdgcn_wmma_f32_16x16x32_f16(false, a, false, bcf[kc], (short)0, cA, false, false);
        }

        /* ---- combine: h_new = u*h + (1-u)*c, mask by seq_len, store y ---- */
        {
            const int n = ncd + nl;
#pragma unroll
            for (int v = 0; v < 8; ++v) {
                int   m  = mb + v;
                float c  = fast_tanh(cA[v]);
                float u  = u_lds[m * F_PITCH + n];
                float h  = h_lds[m * F_PITCH + n];
                float hn = u * h + (1.0f - u) * c;
                bool  valid = (t < slv[v]);
                h_lds[m * F_PITCH + n] = valid ? hn : h;
                out[((size_t)(rowBase + m) * Tc + t) * Hc + n] = valid ? hn : 0.0f;
            }
        }
        /* no trailing barrier: next iteration's top barrier orders h_lds/A reuse */
    }
}

extern "C" void kernel_launch(void* const* d_in, const int* in_sizes, int n_in,
                              void* d_out, int out_size, void* d_ws, size_t ws_size,
                              hipStream_t stream) {
    const float* x   = (const float*)d_in[0];
    const int*   sl  = (const int*)d_in[1];
    const float* Wg  = (const float*)d_in[2];
    const float* bg  = (const float*)d_in[3];
    const float* Wc  = (const float*)d_in[4];
    const float* bc  = (const float*)d_in[5];
    float*       out = (float*)d_out;

    (void)in_sizes; (void)n_in; (void)out_size; (void)d_ws; (void)ws_size;

    /* opt in to >64KB dynamic LDS (320KB/WGP on CDNA5) */
    hipFuncSetAttribute((const void*)dien_gru_kernel,
                        hipFuncAttributeMaxDynamicSharedMemorySize, SMEM_BYTES);

    dim3 grid(Bc / 16);
    dim3 block(NTHREADS);
    dien_gru_kernel<<<grid, block, SMEM_BYTES, stream>>>(x, sl, Wg, bg, Wc, bc, out);
}
